// blstm_cnn_12008728559828
// MI455X (gfx1250) — matrix operand order repaired and checked by reading, not compiled, not run on hardware
//
#include <hip/hip_runtime.h>
#include <hip/hip_bf16.h>
#include <math.h>

// ---------------- model dims ----------------
#define B_   32
#define L_   128
#define W_   16
#define CE_  64
#define NF_  128
#define WE_  256
#define E_   384          // NF_ + WE_
#define H_   512
#define G_   2048         // 4*H
#define M_   (B_*L_)      // 4096 token rows, row = l*32 + b
#define HC_  1024         // 2*H (hcat)
#define NC_  17
#define NCP_ 32           // padded classes
#define GT_  (G_/16)      // 128 gate-column tiles

typedef _Float16 v16h __attribute__((ext_vector_type(16)));
typedef float    v8f  __attribute__((ext_vector_type(8)));
typedef float    v4f  __attribute__((ext_vector_type(4)));

// ---------------- WMMA fragment helpers (wave32, 16x16x32 f16) ----------------
// A: 16x32 (MxK), row-major source X with leading dim ld (halves).
// lane: m = lane&15, g = lane>>4; halves 0..7 = K(8g..8g+7), 8..15 = K(16+8g..).
__device__ inline v16h load_frag_a(const _Float16* X, int ld) {
  int lane = threadIdx.x & 31;
  int m = lane & 15, g = lane >> 4;
  const _Float16* p = X + m * ld + 8 * g;
  v16h a;
  *(v4f*)&a       = *(const v4f*)(p);
  *((v4f*)&a + 1) = *(const v4f*)(p + 16);
  return a;
}
// B: 32x16 (KxN) sourced from a row-major [N,K] weight (PyTorch layout), ld = K.
// lane: n = lane&15, g = lane>>4; halves 0..7 = K(8g..8g+7), 8..15 = K(16+8g..).
__device__ inline v16h load_frag_b(const _Float16* Wnk, int ld) {
  int lane = threadIdx.x & 31;
  int n = lane & 15, g = lane >> 4;
  const _Float16* p = Wnk + n * ld + 8 * g;
  v16h b;
  *(v4f*)&b       = *(const v4f*)(p);
  *((v4f*)&b + 1) = *(const v4f*)(p + 16);
  return b;
}
#define WMMA_F16(a, b, c) \
  __builtin_amdgcn_wmma_f32_16x16x32_f16(false, (a), false, (b), (short)0, (c), false, false)

__device__ inline float sigm(float x) { return 1.f / (1.f + __expf(-x)); }

// ---------------- f32 -> f16 convert (with optional zero-pad tail) ----------------
__global__ void cvt_f16_kernel(const float* __restrict__ src, _Float16* __restrict__ dst,
                               int n, int copy_n) {
  int i = blockIdx.x * 256 + threadIdx.x;
  if (i < n) dst[i] = (i < copy_n) ? (_Float16)src[i] : (_Float16)0.0f;
}

// ---------------- char CNN + embedding gather -> x_h [M_, E_] f16 ----------------
__global__ void char_cnn_embed_kernel(const int* __restrict__ word_ids,
                                      const int* __restrict__ char_ids,
                                      const float* __restrict__ char_emb,
                                      const float* __restrict__ word_emb,
                                      const float* __restrict__ conv_w,
                                      const float* __restrict__ conv_b,
                                      _Float16* __restrict__ x_h) {
  __shared__ float ce[W_][CE_];
  const int blk = blockIdx.x;           // b*L + l
  const int b = blk >> 7, l = blk & 127;
  const int tid = threadIdx.x;          // 128 threads = one filter each
  for (int idx = tid; idx < W_ * CE_; idx += 128) {
    int w = idx >> 6, c = idx & 63;
    int cid = char_ids[blk * W_ + w];
    ce[w][c] = char_emb[cid * CE_ + c];
  }
  __syncthreads();
  const int row = l * B_ + b;
  const int f = tid;
  float acc[W_];
  float bv = conv_b[f];
#pragma unroll
  for (int w = 0; w < W_; ++w) acc[w] = bv;
  for (int c = 0; c < CE_; ++c) {
    float w0 = conv_w[f * 192 + c * 3 + 0];
    float w1 = conv_w[f * 192 + c * 3 + 1];
    float w2 = conv_w[f * 192 + c * 3 + 2];
#pragma unroll
    for (int w = 0; w < W_; ++w) {
      float xm = (w > 0)        ? ce[w - 1][c] : 0.f;
      float x0 = ce[w][c];
      float xp = (w < W_ - 1)   ? ce[w + 1][c] : 0.f;
      acc[w] = fmaf(xm, w0, fmaf(x0, w1, fmaf(xp, w2, acc[w])));
    }
  }
  float m = 0.f;                         // relu(max) == max(relu), lower bound 0
#pragma unroll
  for (int w = 0; w < W_; ++w) m = fmaxf(m, acc[w]);
  x_h[row * E_ + f] = (_Float16)m;
  const int wid = word_ids[b * L_ + l];
  for (int t = tid; t < WE_; t += 128)
    x_h[row * E_ + NF_ + t] = (_Float16)word_emb[wid * WE_ + t];
}

// ---------------- gx = x @ Wih^T + b  (WMMA, C-fragment blocked output) -------
// gx layout: [dir][row-tile mt=0..255][gate-tile gt=0..127][lane][8 floats]
// so the recurrence kernel reloads accumulators with two b128 loads per lane.
__global__ void gates_x_kernel(const _Float16* __restrict__ x_h,
                               const _Float16* __restrict__ wih_h,
                               const float* __restrict__ b_f,
                               const float* __restrict__ b_b,
                               float* __restrict__ gx) {
  const int mt   = blockIdx.x;                    // 0..255  (M tiles)
  const int dir  = blockIdx.z;
  const int wave = threadIdx.x >> 5;              // 0..3
  const int col  = blockIdx.y * 64 + wave * 16;   // gate column tile base
  const _Float16* A0 = x_h + (size_t)mt * 16 * E_;
  const _Float16* Bw = wih_h + (size_t)dir * G_ * E_ + (size_t)col * E_;
  v8f acc = {};
  for (int k = 0; k < E_; k += 32) {
    v16h a  = load_frag_a(A0 + k, E_);
    v16h bf = load_frag_b(Bw + k, E_);
    acc = WMMA_F16(a, bf, acc);
  }
  const int lane = threadIdx.x & 31;
  const int n = lane & 15;
  const float* bias = dir ? b_b : b_f;
  const float bb = bias[col + n];
  v8f o;
#pragma unroll
  for (int r = 0; r < 8; ++r) o[r] = acc[r] + bb;
  float* O = gx + (size_t)dir * M_ * G_
                + ((size_t)mt * GT_ + (col >> 4)) * 256 + lane * 8;
  *(v4f*)O       = *(v4f*)&o;
  *((v4f*)O + 1) = *((v4f*)&o + 1);
}

// ---------------- persistent bidirectional LSTM recurrence ----------------
// 32 WGs: dir = wg>>4, hidden slice = wg&15 (32 h-cols, 128 gate cols each).
// Whh slice (16 cols x 512 K per wave) held in 128 VGPRs for all 128 steps.
// Per step: [32x512]x[512x128] WMMA + gate nonlinearities; cell state in regs;
// h exchanged via L2 ping-pong buffers; device-scope atomic barrier per step.
__device__ inline void grid_sync(unsigned* cnt, unsigned target) {
  __syncthreads();
  if (threadIdx.x == 0) {
    __threadfence();
    atomicAdd(cnt, 1u);
    while (atomicAdd(cnt, 0u) < target) __builtin_amdgcn_s_sleep(1);
  }
  __syncthreads();
  __threadfence();
}

__global__ void __launch_bounds__(256, 1)
bilstm_recurrent_kernel(const float* __restrict__ gx,
                        const _Float16* __restrict__ whh_h,
                        _Float16* __restrict__ hbuf,   // [2 dir][2 pp][32][512] f16
                        _Float16* __restrict__ hs,     // [L][32][1024] f16 (hcat)
                        unsigned* __restrict__ barriers) {
  __shared__ float gates[B_ * 128];                 // 32 rows x 128 local gate cols
  const int wg    = blockIdx.x;
  const int dir   = wg >> 4;
  const int slice = wg & 15;
  const int tid   = threadIdx.x;
  const int lane  = tid & 31;
  const int wave  = tid >> 5;                       // nt = 0..7
  unsigned* cnt = barriers + dir;

  _Float16* hb = hbuf + (size_t)dir * 2 * B_ * H_;
  const _Float16* Wd = whh_h + (size_t)dir * G_ * H_;
  const float* gxd = gx + (size_t)dir * M_ * G_;

  const int chunk = wave >> 1, sub = wave & 1;      // i,f,g,o chunk / 16-col half
  const int gbase = chunk * H_ + slice * 32 + sub * 16;
  const int gt = gbase >> 4;                        // gate-column tile index
  const int n = lane & 15, g2 = lane >> 4;

  // ---- load this wave's Whh slice into registers (lives across all steps) ----
  v16h bfr[16];
#pragma unroll
  for (int kk = 0; kk < 16; ++kk)
    bfr[kk] = load_frag_b(Wd + (size_t)gbase * H_ + kk * 32, H_);

  // zero ping buffer 0 for this slice (h_{-1} = 0)
  for (int i = tid; i < B_ * 32; i += 256) {
    int r = i & 31, j = i >> 5;
    hb[r * H_ + slice * 32 + j] = (_Float16)0.f;
  }
  float cst[4] = {0.f, 0.f, 0.f, 0.f};              // cell state, cell = tid + 256*e

  unsigned epoch = 1;
  grid_sync(cnt, epoch * 16u);                      // h_{-1} visible everywhere

  int p = 0;
  for (int step = 0; step < L_; ++step) {
    const int t = dir ? (L_ - 1 - step) : step;
    const _Float16* Hp = hb + (size_t)p * B_ * H_;

    // warm L2->L0 for next step's gx tiles while this step computes
    if (step + 1 < L_) {
      const int tn = dir ? (L_ - 2 - step) : (step + 1);
      const float* pf = gxd + ((size_t)(tn * 2) * GT_ + gt) * 256 + lane * 8;
      __builtin_prefetch(pf, 0, 1);
      __builtin_prefetch(pf + (size_t)GT_ * 256, 0, 1);
    }

    // init accumulators from precomputed gx[t] (blocked layout, bias folded in)
    v8f acc0, acc1;
    {
      const float* g0 = gxd + ((size_t)(t * 2)     * GT_ + gt) * 256 + lane * 8;
      const float* g1 = gxd + ((size_t)(t * 2 + 1) * GT_ + gt) * 256 + lane * 8;
      *(v4f*)&acc0       = *(const v4f*)g0;
      *((v4f*)&acc0 + 1) = *(const v4f*)(g0 + 4);
      *(v4f*)&acc1       = *(const v4f*)g1;
      *((v4f*)&acc1 + 1) = *(const v4f*)(g1 + 4);
    }
#pragma unroll
    for (int kk = 0; kk < 16; ++kk) {
      v16h a0 = load_frag_a(Hp + kk * 32, H_);
      acc0 = WMMA_F16(a0, bfr[kk], acc0);
      v16h a1 = load_frag_a(Hp + 16 * H_ + kk * 32, H_);
      acc1 = WMMA_F16(a1, bfr[kk], acc1);
    }
    const int lc = wave * 16 + n;                   // local col = chunk*32 + sub*16 + n
#pragma unroll
    for (int r = 0; r < 8; ++r) {
      gates[(8 * g2 + r) * 128 + lc]        = acc0[r];
      gates[(16 + 8 * g2 + r) * 128 + lc]   = acc1[r];
    }
    __syncthreads();
    // ---- elementwise LSTM cell (PyTorch gate order i,f,g,o) ----
    _Float16* Hn = hb + (size_t)(p ^ 1) * B_ * H_;
#pragma unroll
    for (int e = 0; e < 4; ++e) {
      int cell = tid + 256 * e;
      int br = cell & 31, jj = cell >> 5;
      float gi = gates[br * 128 +  0 + jj];
      float gf = gates[br * 128 + 32 + jj];
      float gg = gates[br * 128 + 64 + jj];
      float go = gates[br * 128 + 96 + jj];
      float c  = sigm(gf) * cst[e] + sigm(gi) * tanhf(gg);
      cst[e] = c;
      _Float16 hh = (_Float16)(sigm(go) * tanhf(c));
      Hn[br * H_ + slice * 32 + jj] = hh;
      hs[(size_t)t * B_ * HC_ + (size_t)br * HC_ + dir * H_ + slice * 32 + jj] = hh;
    }
    ++epoch;
    grid_sync(cnt, epoch * 16u);
    p ^= 1;
  }
}

// ---------------- head GEMM1: z = f16(ELU(hcat @ w1^T + b1)) ----------------
__global__ void head1_kernel(const _Float16* __restrict__ hs,
                             const _Float16* __restrict__ w1_h,
                             const float* __restrict__ b1,
                             _Float16* __restrict__ z_h) {
  const int mt   = blockIdx.x;                      // 0..255
  const int wave = threadIdx.x >> 5;
  const int col  = blockIdx.y * 64 + wave * 16;     // 0..511
  const _Float16* A0 = hs + (size_t)mt * 16 * HC_;
  const _Float16* Bw = w1_h + (size_t)col * HC_;
  v8f acc = {};
  for (int k = 0; k < HC_; k += 32) {
    v16h a  = load_frag_a(A0 + k, HC_);
    v16h bf = load_frag_b(Bw + k, HC_);
    acc = WMMA_F16(a, bf, acc);
  }
  const int lane = threadIdx.x & 31;
  const int n = lane & 15, g2 = lane >> 4;
  const float bb = b1[col + n];
#pragma unroll
  for (int r = 0; r < 8; ++r) {
    float v = acc[r] + bb;
    v = v > 0.f ? v : expm1f(v);                    // ELU(alpha=1)
    z_h[(size_t)(mt * 16 + 8 * g2 + r) * 512 + col + n] = (_Float16)v;
  }
}

// ---------------- head GEMM2: out = z @ w2^T + b2 (N padded 17->32) ----------------
__global__ void head2_kernel(const _Float16* __restrict__ z_h,
                             const _Float16* __restrict__ w2_h,
                             const float* __restrict__ b2,
                             float* __restrict__ out) {
  const int mt = blockIdx.x;                        // 0..255
  const int nt = threadIdx.x >> 5;                  // 0..1 (64-thread block)
  const _Float16* A0 = z_h + (size_t)mt * 16 * 512;
  const _Float16* Bw = w2_h + (size_t)nt * 16 * 512;
  v8f acc = {};
  for (int k = 0; k < 512; k += 32) {
    v16h a  = load_frag_a(A0 + k, 512);
    v16h bf = load_frag_b(Bw + k, 512);
    acc = WMMA_F16(a, bf, acc);
  }
  const int lane = threadIdx.x & 31;
  const int n = lane & 15, g2 = lane >> 4;
  const int col = nt * 16 + n;
  if (col < NC_) {
    const float bb = b2[col];
#pragma unroll
    for (int r = 0; r < 8; ++r) {
      int row = mt * 16 + 8 * g2 + r;               // row = l*32 + b
      int b = row & 31, l = row >> 5;
      out[((size_t)b * L_ + l) * NC_ + col] = acc[r] + bb;
    }
  }
}

// ---------------- host side ----------------
extern "C" void kernel_launch(void* const* d_in, const int* in_sizes, int n_in,
                              void* d_out, int out_size, void* d_ws, size_t ws_size,
                              hipStream_t stream) {
  (void)in_sizes; (void)n_in; (void)out_size; (void)ws_size;
  const int*   word_ids = (const int*)  d_in[0];
  const int*   char_ids = (const int*)  d_in[1];
  /* mask d_in[2] unused (all ones; reference ignores it) */
  const float* char_emb = (const float*)d_in[3];
  const float* word_emb = (const float*)d_in[4];
  const float* conv_w   = (const float*)d_in[5];
  const float* conv_b   = (const float*)d_in[6];
  const float* w_ih_f   = (const float*)d_in[7];
  const float* w_hh_f   = (const float*)d_in[8];
  const float* b_f      = (const float*)d_in[9];
  const float* w_ih_b   = (const float*)d_in[10];
  const float* w_hh_b   = (const float*)d_in[11];
  const float* b_b      = (const float*)d_in[12];
  const float* w1       = (const float*)d_in[13];
  const float* b1       = (const float*)d_in[14];
  const float* w2       = (const float*)d_in[15];
  const float* b2       = (const float*)d_in[16];
  float* out = (float*)d_out;

  // workspace carve-out (256B aligned regions)
  char* base = (char*)d_ws;
  size_t off = 0;
  auto carve = [&](size_t bytes) -> char* {
    char* p = base + off;
    off = (off + bytes + 255) & ~(size_t)255;
    return p;
  };
  unsigned*  barriers = (unsigned*) carve(256);
  _Float16*  x_h      = (_Float16*)carve((size_t)M_ * E_ * 2);
  _Float16*  wih_h    = (_Float16*)carve((size_t)2 * G_ * E_ * 2);
  _Float16*  whh_h    = (_Float16*)carve((size_t)2 * G_ * H_ * 2);
  _Float16*  w1_h     = (_Float16*)carve((size_t)512 * HC_ * 2);
  _Float16*  w2_h     = (_Float16*)carve((size_t)NCP_ * 512 * 2);
  float*     gx       = (float*)   carve((size_t)2 * M_ * G_ * 4);
  _Float16*  hbuf     = (_Float16*)carve((size_t)2 * 2 * B_ * H_ * 2);
  _Float16*  hs       = (_Float16*)carve((size_t)M_ * HC_ * 2);
  _Float16*  z_h      = (_Float16*)carve((size_t)M_ * 512 * 2);

  hipMemsetAsync(barriers, 0, 256, stream);        // reset grid-barrier counters

  // weight converts f32 -> f16
  {
    int n;
    n = G_ * E_;
    cvt_f16_kernel<<<(n + 255) / 256, 256, 0, stream>>>(w_ih_f, wih_h,          n, n);
    cvt_f16_kernel<<<(n + 255) / 256, 256, 0, stream>>>(w_ih_b, wih_h + n,      n, n);
    n = G_ * H_;
    cvt_f16_kernel<<<(n + 255) / 256, 256, 0, stream>>>(w_hh_f, whh_h,          n, n);
    cvt_f16_kernel<<<(n + 255) / 256, 256, 0, stream>>>(w_hh_b, whh_h + n,      n, n);
    n = 512 * HC_;
    cvt_f16_kernel<<<(n + 255) / 256, 256, 0, stream>>>(w1, w1_h, n, n);
    cvt_f16_kernel<<<(NCP_ * 512 + 255) / 256, 256, 0, stream>>>(w2, w2_h,
                                                                 NCP_ * 512, NC_ * 512);
  }
  // char CNN + embeddings -> x_h
  char_cnn_embed_kernel<<<M_, 128, 0, stream>>>(word_ids, char_ids, char_emb, word_emb,
                                                conv_w, conv_b, x_h);
  // input projections (both directions)
  gates_x_kernel<<<dim3(M_ / 16, G_ / 64, 2), 128, 0, stream>>>(x_h, wih_h, b_f, b_b, gx);
  // persistent recurrence (32 co-resident WGs, 16 per direction)
  bilstm_recurrent_kernel<<<32, 256, 0, stream>>>(gx, whh_h, hbuf, hs, barriers);
  // head
  head1_kernel<<<dim3(M_ / 16, 512 / 64), 128, 0, stream>>>(hs, w1_h, b1, z_h);
  head2_kernel<<<M_ / 16, 64, 0, stream>>>(z_h, w2_h, b2, out);
}
